// EfficientDet_11982958756213
// MI455X (gfx1250) — matrix operand order repaired and checked by reading, not compiled, not run on hardware
//
#include <hip/hip_runtime.h>
#include <hip/hip_bf16.h>

typedef __attribute__((ext_vector_type(16))) _Float16 v16h;
typedef __attribute__((ext_vector_type(8)))  _Float16 v8h;
typedef __attribute__((ext_vector_type(8)))  float    v8f;
typedef __attribute__((ext_vector_type(4)))  int      v4i;

#define CCH 64   // BiFPN channels
#define BB  8    // batch

#if defined(__AMDGCN__) && __has_builtin(__builtin_amdgcn_global_load_async_to_lds_b128)
#define HAVE_ASYNC_LDS 1
#else
#define HAVE_ASYNC_LDS 0
#endif

typedef __attribute__((address_space(1))) v4i v4i_g;   // global
typedef __attribute__((address_space(3))) v4i v4i_l;   // LDS

// fast reciprocal (v_rcp_f32) -- output feeds an fp16 store, full IEEE divide
// expansion (div_scale/rcp/fma chain) is wasted VALU work.
__device__ __forceinline__ float fast_rcp(float x) {
#if defined(__AMDGCN__) && __has_builtin(__builtin_amdgcn_rcpf)
  return __builtin_amdgcn_rcpf(x);
#else
  return 1.f / x;
#endif
}

// ---------------------------------------------------------------------------
// Stage one contiguous 2KB B-tile (16 pixels x 64ch fp16) into LDS.
// 128 threads x 16B. Uses CDNA5 async load-to-LDS when available.
// ---------------------------------------------------------------------------
__device__ __forceinline__ void stage_tile(const _Float16* __restrict__ g,
                                           _Float16* lds) {
  int t = (int)threadIdx.x;
  if (t < 128) {
#if HAVE_ASYNC_LDS
    __builtin_amdgcn_global_load_async_to_lds_b128(
        (v4i_g*)(g + t * 8), (v4i_l*)(lds + t * 8), 0, 0);
#else
    ((int4*)lds)[t] = ((const int4*)g)[t];
#endif
  }
#if HAVE_ASYNC_LDS
#if __has_builtin(__builtin_amdgcn_s_wait_asynccnt)
  __builtin_amdgcn_s_wait_asynccnt(0);
#else
  asm volatile("s_wait_asynccnt 0" ::: "memory");
#endif
#endif
  __syncthreads();
}

// ---------------------------------------------------------------------------
// Convert NCHW fp32 feature map -> [p][c] fp16 (c contiguous), p = n*HW+h*W+w
// ---------------------------------------------------------------------------
__global__ void cvt_nchw_to_pc(const float* __restrict__ in,
                               _Float16* __restrict__ out,
                               int HW, int total) {
  int idx = blockIdx.x * 256 + threadIdx.x;
  if (idx >= total) return;
  int c = idx & 63;
  int p = idx >> 6;
  int n = p / HW;
  int hw = p - n * HW;
  out[idx] = (_Float16)in[(n * CCH + c) * HW + hw];
}

// ---------------------------------------------------------------------------
// fp32 [Mreal][64] weights -> fp16 [Mpad][64], zero rows >= Mreal.
// ---------------------------------------------------------------------------
__global__ void cvt_w_f16(const float* __restrict__ in,
                          _Float16* __restrict__ out,
                          int Mreal, int total) {
  int idx = blockIdx.x * 256 + threadIdx.x;
  if (idx >= total) return;
  int m = idx >> 6;
  out[idx] = (m < Mreal) ? (_Float16)in[idx] : (_Float16)0.f;
}

// ---------------------------------------------------------------------------
// Fuse conv-bias + BN into per-channel scale/shift:
//   sA = g*rsqrt(v+eps);  sB = (pw_b - m)*sA + b
// idx = (lvl*3+d)*64 + o over (5,3,64)
// ---------------------------------------------------------------------------
__global__ void bn_prep(const float* __restrict__ g, const float* __restrict__ b,
                        const float* __restrict__ m, const float* __restrict__ v,
                        const float* __restrict__ pwb,  // (3,64)
                        float* __restrict__ sA, float* __restrict__ sB, int total) {
  int idx = blockIdx.x * 256 + threadIdx.x;
  if (idx >= total) return;
  int o = idx & 63;
  int d = (idx >> 6) % 3;
  float s = g[idx] * rsqrtf(v[idx] + 1e-3f);
  sA[idx] = s;
  sB[idx] = (pwb[d * 64 + o] - m[idx]) * s + b[idx];
}

// ---------------------------------------------------------------------------
// Depthwise 3x3, SAME padding, fp16 [p][c] -> fp16 [p][c]; weights fp32 [C][9]
// ---------------------------------------------------------------------------
__global__ void dw3x3(const _Float16* __restrict__ X,
                      const float* __restrict__ Wdw,
                      _Float16* __restrict__ Y,
                      int H, int W, int total) {
  int idx = blockIdx.x * 256 + threadIdx.x;
  if (idx >= total) return;
  int c = idx & 63;
  int p = idx >> 6;
  int HW = H * W;
  int n  = p / HW;
  int hw = p - n * HW;
  int h  = hw / W;
  int w  = hw - h * W;
  const float* wc = Wdw + c * 9;
  float acc = 0.f;
#pragma unroll
  for (int kh = 0; kh < 3; ++kh) {
    int hh = h + kh - 1;
    if (hh < 0 || hh >= H) continue;
#pragma unroll
    for (int kw = 0; kw < 3; ++kw) {
      int ww = w + kw - 1;
      if (ww < 0 || ww >= W) continue;
      acc += wc[kh * 3 + kw] * (float)X[((n * HW + hh * W + ww) << 6) + c];
    }
  }
  Y[idx] = (_Float16)acc;
}

// ---------------------------------------------------------------------------
// Shared WMMA mainloop: one 16x16 (M,N) tile per wave, K=64 in 2 steps.
// A from fp16 weights in global (L2-hot), B from the staged LDS tile.
// ---------------------------------------------------------------------------
__device__ __forceinline__ v8f wmma_tile(const _Float16* __restrict__ wrow,
                                         const _Float16* lds,
                                         int half, int l15) {
  v8f acc = {};
#pragma unroll
  for (int kt = 0; kt < CCH; kt += 32) {
    v16h a = *(const v16h*)(wrow + kt + half * 16);        // K = kt+half*16+0..15
    v16h b = *(const v16h*)(lds + l15 * CCH + half * 16 + kt);  // K = kt+half*16+0..15
    acc = __builtin_amdgcn_wmma_f32_16x16x32_f16(
        false, a, false, b, (short)0, acc, false, false);
  }
  return acc;
}

// ---------------------------------------------------------------------------
// Subnet pointwise GEMM (M=64) + fused BN scale/shift + swish -> fp16 [p][64]
// Block = 128 threads (4 waves), wave w owns M-tile w; all share N-tile blockIdx.x
// ---------------------------------------------------------------------------
__global__ void pw_gemm_bn_swish(const _Float16* __restrict__ X,
                                 const _Float16* __restrict__ Wh,   // fp16 [64][64]
                                 const float* __restrict__ sA,
                                 const float* __restrict__ sB,
                                 _Float16* __restrict__ Y) {
  __shared__ __align__(32) _Float16 lds[16 * CCH];
  int nt = (int)blockIdx.x;
  stage_tile(X + (size_t)nt * 16 * CCH, lds);

  int mt   = (int)(threadIdx.x >> 5);       // 4 waves = 4 M-tiles
  int lane = (int)(threadIdx.x & 31);
  int half = lane >> 4;
  int l15  = lane & 15;

  v8f acc = wmma_tile(Wh + (mt * 16 + l15) * CCH, lds, half, l15);

  int ob = mt * 16 + half * 8;              // D: VGPR r -> channel ob + r
  v8f sav = *(const v8f*)(sA + ob);
  v8f sbv = *(const v8f*)(sB + ob);
  v8h outh;
#pragma unroll
  for (int r = 0; r < 8; ++r) {
    float y = acc[r] * sav[r] + sbv[r];
    y = y * fast_rcp(1.f + __expf(-y));     // swish via v_rcp_f32
    outh[r] = (_Float16)y;
  }
  *(v8h*)(Y + ((size_t)nt * 16 + l15) * CCH + ob) = outh;
}

// ---------------------------------------------------------------------------
// Head pointwise GEMM: fp16 weights padded to 16*Mtiles rows; + bias;
// fp32 output scattered into NCHW [n][o][h][w].
// Block = 256 threads (8 waves); grid = (Ntiles, ceil(Mtiles/8)).
// ---------------------------------------------------------------------------
__global__ void pw_gemm_head(const _Float16* __restrict__ X,
                             const _Float16* __restrict__ Wh,
                             const float* __restrict__ bias,
                             float* __restrict__ out,
                             int Mreal, int Mtiles, int HW) {
  __shared__ __align__(32) _Float16 lds[16 * CCH];
  int nt = (int)blockIdx.x;
  stage_tile(X + (size_t)nt * 16 * CCH, lds);

  int mt = (int)blockIdx.y * (int)(blockDim.x >> 5) + (int)(threadIdx.x >> 5);
  if (mt >= Mtiles) return;                 // uniform per wave, after barrier
  int lane = (int)(threadIdx.x & 31);
  int half = lane >> 4;
  int l15  = lane & 15;

  v8f acc = wmma_tile(Wh + (mt * 16 + l15) * CCH, lds, half, l15);

  int p  = nt * 16 + l15;
  int n  = p / HW;
  int hw = p - n * HW;
#pragma unroll
  for (int r = 0; r < 8; ++r) {
    int o = mt * 16 + half * 8 + r;
    if (o < Mreal)
      out[(n * Mreal + o) * HW + hw] = acc[r] + bias[o];
  }
}

// ---------------------------------------------------------------------------
extern "C" void kernel_launch(void* const* d_in, const int* in_sizes, int n_in,
                              void* d_out, int out_size, void* d_ws, size_t ws_size,
                              hipStream_t stream) {
  (void)in_sizes; (void)n_in; (void)out_size; (void)ws_size;

  const float* feats[5] = {(const float*)d_in[0], (const float*)d_in[1],
                           (const float*)d_in[2], (const float*)d_in[3],
                           (const float*)d_in[4]};
  const float* sub_dw[2]  = {(const float*)d_in[5],  (const float*)d_in[8]};   // cls, box
  const float* sub_pw[2]  = {(const float*)d_in[6],  (const float*)d_in[9]};
  const float* sub_pwb[2] = {(const float*)d_in[7],  (const float*)d_in[10]};
  const float* bn_g[2]    = {(const float*)d_in[11], (const float*)d_in[15]};
  const float* bn_b[2]    = {(const float*)d_in[12], (const float*)d_in[16]};
  const float* bn_m[2]    = {(const float*)d_in[13], (const float*)d_in[17]};
  const float* bn_v[2]    = {(const float*)d_in[14], (const float*)d_in[18]};
  const float* head_dw[2] = {(const float*)d_in[19], (const float*)d_in[22]};  // score, pred
  const float* head_pw[2] = {(const float*)d_in[20], (const float*)d_in[23]};
  const float* head_b[2]  = {(const float*)d_in[21], (const float*)d_in[24]};
  const int    head_M[2]  = {810, 36};
  const int    head_MT[2] = {51, 3};

  float* out = (float*)d_out;

  // ---- scratch layout (bytes) ----
  const size_t BUF_HALVES = (size_t)BB * 64 * 64 * CCH;   // 2 Mi halves
  char* ws = (char*)d_ws;
  _Float16* buf0   = (_Float16*)ws;                            // 4 MiB
  _Float16* buf1   = buf0 + BUF_HALVES;                        // 4 MiB
  _Float16* wsub16 = buf1 + BUF_HALVES;                        // 2*3*64*64 halves
  _Float16* whead16 = wsub16 + 2 * 3 * 64 * 64;                // 816*64 + 48*64 halves
  float* sAb = (float*)(whead16 + 816 * 64 + 48 * 64);         // 2*960 floats
  float* sBb = sAb + 2 * 960;

  // ---- one-time preprocessing (cheap, graph-capturable) ----
  cvt_w_f16<<<(2 * 3 * 64 * 64 + 255) / 256, 256, 0, stream>>>(
      sub_pw[0], wsub16, 3 * 64, 3 * 64 * 64);                 // cls (no padding)
  cvt_w_f16<<<(3 * 64 * 64 + 255) / 256, 256, 0, stream>>>(
      sub_pw[1], wsub16 + 3 * 64 * 64, 3 * 64, 3 * 64 * 64);   // box
  cvt_w_f16<<<(816 * 64 + 255) / 256, 256, 0, stream>>>(
      head_pw[0], whead16, 810, 816 * 64);                     // score, pad 810->816
  cvt_w_f16<<<(48 * 64 + 255) / 256, 256, 0, stream>>>(
      head_pw[1], whead16 + 816 * 64, 36, 48 * 64);            // pred, pad 36->48
  for (int s = 0; s < 2; ++s)
    bn_prep<<<(960 + 255) / 256, 256, 0, stream>>>(
        bn_g[s], bn_b[s], bn_m[s], bn_v[s], sub_pwb[s],
        sAb + s * 960, sBb + s * 960, 960);

  // ---- output offsets ----
  const int sizes[5] = {64, 32, 16, 8, 4};
  long head_off[2][5];
  {
    long off = 0;
    for (int h = 0; h < 2; ++h)
      for (int l = 0; l < 5; ++l) {
        head_off[h][l] = off;
        off += (long)BB * head_M[h] * sizes[l] * sizes[l];
      }
  }

  for (int lvl = 0; lvl < 5; ++lvl) {
    const int H = sizes[lvl], W = H, HW = H * W;
    const int P = BB * HW;
    const int totalPC = P * CCH;
    const int blkPC = (totalPC + 255) / 256;
    const int Ntiles = P / 16;

    for (int s = 0; s < 2; ++s) {             // s=0: cls/score, s=1: box/pred
      cvt_nchw_to_pc<<<blkPC, 256, 0, stream>>>(feats[lvl], buf0, HW, totalPC);
      for (int d = 0; d < 3; ++d) {
        dw3x3<<<blkPC, 256, 0, stream>>>(buf0, sub_dw[s] + d * CCH * 9,
                                         buf1, H, W, totalPC);
        pw_gemm_bn_swish<<<Ntiles, 128, 0, stream>>>(
            buf1, wsub16 + (s * 3 + d) * CCH * CCH,
            sAb + s * 960 + (lvl * 3 + d) * 64,
            sBb + s * 960 + (lvl * 3 + d) * 64, buf0);
      }
      dw3x3<<<blkPC, 256, 0, stream>>>(buf0, head_dw[s], buf1, H, W, totalPC);
      dim3 g(Ntiles, (head_MT[s] + 7) / 8);
      pw_gemm_head<<<g, 256, 0, stream>>>(
          buf1, whead16 + (s ? 816 * 64 : 0), head_b[s],
          out + head_off[s][lvl], head_M[s], head_MT[s], HW);
    }
  }
}
